// MultiHeadAttention_74904229642815
// MI455X (gfx1250) — matrix operand order repaired and checked by reading, not compiled, not run on hardware
//
#include <hip/hip_runtime.h>

// ---------------- types ----------------
typedef __attribute__((ext_vector_type(16))) __bf16 v16bf;
typedef __attribute__((ext_vector_type(8)))  __bf16 v8bf;
typedef __attribute__((ext_vector_type(8)))  float  v8f;

#define B_      2
#define T_      2048
#define E_      1024
#define H_      16
#define D_      64
#define THREE_E 3072

#define WMMA_BF16(a, b, c) \
  __builtin_amdgcn_wmma_f32_16x16x32_bf16(false, (a), false, (b), (short)0, (c), false, false)

static __device__ __forceinline__ v8f vzero8() {
  v8f z;
#pragma unroll
  for (int i = 0; i < 8; ++i) z[i] = 0.0f;
  return z;
}

// A fragment (16x32 bf16, M x K). Lane holds row (lane&15); two 16B chunks:
//   halves 0..7  -> K = k0 + (lane<16 ? 0 : 8)  .. +8
//   halves 8..15 -> K = k0 + 16 + (lane<16 ? 0 : 8) .. +8
static __device__ __forceinline__ v16bf load_a_frag(const __bf16* base, int ld, int row,
                                                    int k0, int lane) {
  const __bf16* p = base + (size_t)row * ld + k0;
  const int g = (lane & 16) ? 8 : 0;
  union { v16bf v; v8bf h[2]; } u;
  u.h[0] = *(const v8bf*)(p + g);
  u.h[1] = *(const v8bf*)(p + 16 + g);
  return u.v;
}

// B fragment (32x16 bf16, K x N) from K-contiguous storage: element (k,n) at
// base[n*ld + k]. Lane holds col (lane&15); halves 0..7 -> K = k0 + (lane<16?0:8) + i,
//   halves 8..15 -> K = k0 + 16 + (lane<16?0:8) + (i-8).
static __device__ __forceinline__ v16bf load_b_frag(const __bf16* base, int ld, int col,
                                                    int k0, int lane) {
  const __bf16* p = base + (size_t)col * ld + k0 + ((lane & 16) ? 8 : 0);
  union { v16bf v; v8bf h[2]; } u;
  u.h[0] = *(const v8bf*)(p);
  u.h[1] = *(const v8bf*)(p + 16);
  return u.v;
}

// ---------------- conversion kernels ----------------
__global__ void k_cvt_bf16(const float* __restrict__ in, __bf16* __restrict__ out, int n) {
  int i = blockIdx.x * blockDim.x + threadIdx.x;
  if (i < n) out[i] = (__bf16)in[i];
}

// in [rows][cols] (f32 row-major) -> out [cols][rows] (bf16)
__global__ void k_cvt_transpose_bf16(const float* __restrict__ in, __bf16* __restrict__ out,
                                     int rows, int cols) {
  int i = blockIdx.x * blockDim.x + threadIdx.x;
  if (i >= rows * cols) return;
  int c = i / rows;  // output row (= input col)
  int r = i % rows;  // output col (= input row)
  out[(size_t)c * rows + r] = (__bf16)in[(size_t)r * cols + c];
}

// ---------------- QKV projection ----------------
// C[4096,3072] = Xb[4096,1024] @ Wqkv + b ; scatter into Q[B,H,T,D], K[B,H,T,D], V^T[B,H,D,T]
// Per-wave tile: 32(M) x 64(N) -> 8 WMMAs / 32-deep k-step, each B-frag used twice.
__global__ __launch_bounds__(256) void k_gemm_qkv(const __bf16* __restrict__ Xb,
                                                  const __bf16* __restrict__ WT,
                                                  const float* __restrict__ bias,
                                                  __bf16* __restrict__ Qb,
                                                  __bf16* __restrict__ Kb,
                                                  __bf16* __restrict__ VTb) {
  const int lane   = threadIdx.x & 31;
  const int wave   = threadIdx.x >> 5;
  const int task   = blockIdx.x * 8 + wave;  // 128 M-tiles * 48 N-supertiles = 6144
  const int NSUP   = THREE_E / 64;           // 48
  const int mTile  = task / NSUP;
  const int nSup   = task % NSUP;
  const int lane16 = lane & 15;
  const int mb     = mTile * 32;
  const int nb     = nSup * 64;

  v8f acc[2][4];
#pragma unroll
  for (int mi = 0; mi < 2; ++mi)
#pragma unroll
    for (int j = 0; j < 4; ++j) acc[mi][j] = vzero8();

  for (int k0 = 0; k0 < E_; k0 += 32) {
    v16bf a0 = load_a_frag(Xb, E_, mb + lane16, k0, lane);
    v16bf a1 = load_a_frag(Xb, E_, mb + 16 + lane16, k0, lane);
#pragma unroll
    for (int j = 0; j < 4; ++j) {
      v16bf b = load_b_frag(WT, E_, nb + j * 16 + lane16, k0, lane);
      acc[0][j] = WMMA_BF16(a0, b, acc[0][j]);
      acc[1][j] = WMMA_BF16(a1, b, acc[1][j]);
    }
  }

  const int hi8 = (lane & 16) ? 8 : 0;
#pragma unroll
  for (int j = 0; j < 4; ++j) {
    const int   c     = nb + j * 16 + lane16;
    const float bv    = bias[c];
    const int   which = c / E_;  // 0=q 1=k 2=v
    const int   e     = c % E_;
    const int   h     = e / D_;
    const int   d     = e % D_;
#pragma unroll
    for (int mi = 0; mi < 2; ++mi) {
#pragma unroll
      for (int r = 0; r < 8; ++r) {
        const int    R   = mb + mi * 16 + hi8 + r;
        const int    bb  = R / T_;
        const int    t   = R % T_;
        const __bf16 vb  = (__bf16)(acc[mi][j][r] + bv);
        const size_t bht = (size_t)(bb * H_ + h);
        if (which == 0)      Qb[(bht * T_ + t) * D_ + d] = vb;
        else if (which == 1) Kb[(bht * T_ + t) * D_ + d] = vb;
        else                 VTb[(bht * D_ + d) * T_ + t] = vb;
      }
    }
  }
}

// ---------------- flash attention (causal) ----------------
// One wave per (b,h, 16-query tile); online softmax over 32-key blocks.
__global__ __launch_bounds__(256) void k_attn(const __bf16* __restrict__ Qb,
                                              const __bf16* __restrict__ Kb,
                                              const __bf16* __restrict__ VTb,
                                              __bf16* __restrict__ Ob) {
  __shared__ __bf16 Plds[8 * 16 * 32];  // 1KB per wave: P tile staging (C->A relayout)
  const int lane   = threadIdx.x & 31;
  const int wave   = threadIdx.x >> 5;
  const int lane16 = lane & 15;
  const int hi8    = (lane & 16) ? 8 : 0;
  const int task   = blockIdx.x * 8 + wave;  // B*H*(T/16) = 4096
  const int bh     = task >> 7;              // / 128 q-tiles
  const int qt     = task & 127;
  const int qbase  = qt * 16;

  const __bf16* Qh = Qb + (size_t)bh * T_ * D_;
  const __bf16* Kh = Kb + (size_t)bh * T_ * D_;
  const __bf16* Vh = VTb + (size_t)bh * D_ * T_;
  __bf16*       Pw = Plds + wave * (16 * 32);

  const v16bf aq0 = load_a_frag(Qh, D_, qbase + lane16, 0, lane);
  const v16bf aq1 = load_a_frag(Qh, D_, qbase + lane16, 32, lane);

  v8f o[4];
#pragma unroll
  for (int j = 0; j < 4; ++j) o[j] = vzero8();
  float rmax[8], rsum[8];
#pragma unroll
  for (int r = 0; r < 8; ++r) { rmax[r] = -1e30f; rsum[r] = 0.0f; }

  const float sc = 0.125f;  // 1/sqrt(64)

  for (int kb = 0; kb < qbase + 16; kb += 32) {
    // S = Q K^T for two 16-wide key tiles (K rows are contiguous d => direct B-frag)
    v8f s0 = vzero8(), s1 = vzero8();
    s0 = WMMA_BF16(aq0, load_b_frag(Kh, D_, kb + lane16, 0, lane), s0);
    s0 = WMMA_BF16(aq1, load_b_frag(Kh, D_, kb + lane16, 32, lane), s0);
    s1 = WMMA_BF16(aq0, load_b_frag(Kh, D_, kb + 16 + lane16, 0, lane), s1);
    s1 = WMMA_BF16(aq1, load_b_frag(Kh, D_, kb + 16 + lane16, 32, lane), s1);

    const int n0 = kb + lane16;
    const int n1 = kb + 16 + lane16;
    float p0v[8], p1v[8], oscale[8];
#pragma unroll
    for (int r = 0; r < 8; ++r) {
      const int qi = qbase + hi8 + r;
      float v0 = (n0 <= qi) ? s0[r] * sc : -1e30f;  // causal: key <= query
      float v1 = (n1 <= qi) ? s1[r] * sc : -1e30f;
      float bm = fmaxf(v0, v1);
#pragma unroll
      for (int off = 1; off < 16; off <<= 1)  // reduce over 16-lane half (rows stay put)
        bm = fmaxf(bm, __shfl_xor(bm, off, 32));
      const float nm  = fmaxf(rmax[r], bm);
      const float scl = __expf(rmax[r] - nm);
      const float p0  = __expf(v0 - nm);
      const float p1  = __expf(v1 - nm);
      float bs = p0 + p1;
#pragma unroll
      for (int off = 1; off < 16; off <<= 1) bs += __shfl_xor(bs, off, 32);
      rsum[r]   = rsum[r] * scl + bs;
      rmax[r]   = nm;
      oscale[r] = scl;
      p0v[r]    = p0;
      p1v[r]    = p1;
    }
#pragma unroll
    for (int j = 0; j < 4; ++j)
#pragma unroll
      for (int r = 0; r < 8; ++r) o[j][r] *= oscale[r];

    // C-layout f32 -> A-layout bf16 via per-wave LDS tile (16x32)
#pragma unroll
    for (int r = 0; r < 8; ++r) {
      const int m = hi8 + r;
      Pw[m * 32 + lane16]      = (__bf16)p0v[r];
      Pw[m * 32 + 16 + lane16] = (__bf16)p1v[r];
    }
    asm volatile("s_wait_dscnt 0x0" ::: "memory");  // same-wave cross-lane LDS RAW
    const v16bf ap = load_a_frag(Pw, 32, lane16, 0, lane);

    // O += P @ V  (V^T rows are key-contiguous => direct B-frag)
#pragma unroll
    for (int j = 0; j < 4; ++j) {
      v16bf bv = load_b_frag(Vh, T_, j * 16 + lane16, kb, lane);
      o[j] = WMMA_BF16(ap, bv, o[j]);
    }
  }

  // normalize + store merged heads: Ob[b, t, h*64 + d] (bf16)
  const int b = bh / H_, h = bh % H_;
#pragma unroll
  for (int r = 0; r < 8; ++r) {
    const float  inv    = 1.0f / rsum[r];
    const int    t      = qbase + hi8 + r;
    const size_t rowoff = ((size_t)(b * T_ + t)) * E_ + h * D_;
#pragma unroll
    for (int j = 0; j < 4; ++j)
      Ob[rowoff + j * 16 + lane16] = (__bf16)(o[j][r] * inv);
  }
}

// ---------------- output projection ----------------
// Out[4096,1024] (f32) = Ob[4096,1024] @ W_out + b_out ; 32x64 per-wave tile.
__global__ __launch_bounds__(256) void k_gemm_out(const __bf16* __restrict__ Ab,
                                                  const __bf16* __restrict__ WT,
                                                  const float* __restrict__ bias,
                                                  float* __restrict__ Out) {
  const int lane   = threadIdx.x & 31;
  const int wave   = threadIdx.x >> 5;
  const int task   = blockIdx.x * 8 + wave;  // 128 * 16 = 2048
  const int NSUP   = E_ / 64;                // 16
  const int mTile  = task / NSUP;
  const int nSup   = task % NSUP;
  const int lane16 = lane & 15;
  const int mb     = mTile * 32;
  const int nb     = nSup * 64;

  v8f acc[2][4];
#pragma unroll
  for (int mi = 0; mi < 2; ++mi)
#pragma unroll
    for (int j = 0; j < 4; ++j) acc[mi][j] = vzero8();

  for (int k0 = 0; k0 < E_; k0 += 32) {
    v16bf a0 = load_a_frag(Ab, E_, mb + lane16, k0, lane);
    v16bf a1 = load_a_frag(Ab, E_, mb + 16 + lane16, k0, lane);
#pragma unroll
    for (int j = 0; j < 4; ++j) {
      v16bf b = load_b_frag(WT, E_, nb + j * 16 + lane16, k0, lane);
      acc[0][j] = WMMA_BF16(a0, b, acc[0][j]);
      acc[1][j] = WMMA_BF16(a1, b, acc[1][j]);
    }
  }

  const int hi8 = (lane & 16) ? 8 : 0;
#pragma unroll
  for (int j = 0; j < 4; ++j) {
    const int   c  = nb + j * 16 + lane16;
    const float bv = bias[c];
#pragma unroll
    for (int mi = 0; mi < 2; ++mi) {
#pragma unroll
      for (int r = 0; r < 8; ++r) {
        const int R = mb + mi * 16 + hi8 + r;
        Out[(size_t)R * E_ + c] = acc[mi][j][r] + bv;
      }
    }
  }
}

// ---------------- launcher ----------------
extern "C" void kernel_launch(void* const* d_in, const int* in_sizes, int n_in,
                              void* d_out, int out_size, void* d_ws, size_t ws_size,
                              hipStream_t stream) {
  const float* x     = (const float*)d_in[0];
  // d_in[1] = mask (causal, reconstructed analytically -> unused)
  const float* W_qkv = (const float*)d_in[2];
  const float* b_qkv = (const float*)d_in[3];
  const float* W_out = (const float*)d_in[4];
  const float* b_out = (const float*)d_in[5];
  float*       out   = (float*)d_out;

  char* ws = (char*)d_ws;
  __bf16* xb  = (__bf16*)(ws + 0);          //  8 MB  x bf16 [4096,1024]
  __bf16* WqT = (__bf16*)(ws + 8388608);    //  6 MB  W_qkv^T bf16 [3072,1024]
  __bf16* WoT = (__bf16*)(ws + 14680064);   //  2 MB  W_out^T bf16 [1024,1024]
  __bf16* Qb  = (__bf16*)(ws + 16777216);   //  8 MB  Q [B,H,T,D]
  __bf16* Kb  = (__bf16*)(ws + 25165824);   //  8 MB  K [B,H,T,D]
  __bf16* VTb = (__bf16*)(ws + 33554432);   //  8 MB  V^T [B,H,D,T]
  __bf16* Ob  = (__bf16*)(ws + 41943040);   //  8 MB  attn out [B,T,E]

  const int nx = B_ * T_ * E_;
  k_cvt_bf16<<<(nx + 255) / 256, 256, 0, stream>>>(x, xb, nx);
  k_cvt_transpose_bf16<<<(E_ * THREE_E + 255) / 256, 256, 0, stream>>>(W_qkv, WqT, E_, THREE_E);
  k_cvt_transpose_bf16<<<(E_ * E_ + 255) / 256, 256, 0, stream>>>(W_out, WoT, E_, E_);

  k_gemm_qkv<<<768, 256, 0, stream>>>(xb, WqT, b_qkv, Qb, Kb, VTb);  // 6144 waves
  k_attn<<<512, 256, 0, stream>>>(Qb, Kb, VTb, Ob);                  // 4096 waves
  k_gemm_out<<<256, 256, 0, stream>>>(Ob, WoT, b_out, out);          // 2048 waves
}
